// TransformerBlock_56032143344237
// MI455X (gfx1250) — matrix operand order repaired and checked by reading, not compiled, not run on hardware
//
#include <hip/hip_runtime.h>

// ---------------------------------------------------------------------------
// MI455X (gfx1250) transformer block.
//  * All GEMMs + attention on v_wmma_f32_16x16x32_bf16 (wave32).
//  * GEMM LDS tiles are filled by the Tensor Data Mover (tensor_load_to_lds,
//    TENSORcnt double-buffered) -> DMA overlaps WMMA issue.
//  * bf16 activations/weights, f32 accumulation. ~135 GFLOP matrix work vs
//    ~180MB HBM traffic -> matrix-op bound, so everything K>=64 uses WMMA.
// ---------------------------------------------------------------------------

#define B_   8
#define TQ_  512
#define TK_  2048
#define D_   1024
#define NA_  512
#define NI_  1536
#define H_   16
#define HD_  64
#define NE_  4
#define HID_ 32

typedef __bf16 bf16;
typedef __attribute__((ext_vector_type(16))) __bf16 v16bf;
typedef __attribute__((ext_vector_type(8)))  __bf16 bf16x8;
typedef __attribute__((ext_vector_type(8)))  float  v8f;
typedef __attribute__((ext_vector_type(4)))  unsigned int u32x4;
typedef __attribute__((ext_vector_type(8)))  int i32x8;
typedef __attribute__((ext_vector_type(4)))  int i32x4;

__device__ __forceinline__ v16bf make_frag(const bf16* p0, const bf16* p1) {
  bf16x8 lo = *(const bf16x8*)p0;
  bf16x8 hi = *(const bf16x8*)p1;
  return __builtin_shufflevector(lo, hi, 0,1,2,3,4,5,6,7,8,9,10,11,12,13,14,15);
}

__device__ __forceinline__ float sigmoidf_(float x) {
  return 1.0f / (1.0f + __expf(-x));
}

// ---------------------------------------------------------------------------
// Tensor Data Mover: 2D tile load Global->LDS (CDNA5 ISA ch.8 D# layout).
//  g0: [1:0]=count=1 | lds_addr | global_addr[56:0] | type=2
//  g1: data_size=1(2B), tensor_dim0/1, tile_dim0/1, tensor_dim0_stride
// Issued by ONE wave only (TDM ignores EXEC); completion via TENSORcnt.
// ---------------------------------------------------------------------------
__device__ __forceinline__ void tdm_load_2d(unsigned int lds_off, const void* gptr,
                                            unsigned int tensor_d0, unsigned int tensor_d1,
                                            unsigned int tile_d0,  unsigned int tile_d1,
                                            unsigned long long stride0_elems)
{
  unsigned long long ga = (unsigned long long)gptr;
  u32x4 g0;
  g0[0] = 1u;                                            // count=1 user D#
  g0[1] = lds_off;                                       // LDS byte address
  g0[2] = (unsigned int)(ga & 0xffffffffu);              // global_addr[31:0]
  g0[3] = (unsigned int)((ga >> 32) & 0x01ffffffu)       // global_addr[56:32]
        | (2u << 30);                                    // type = 2 ("image")
  i32x8 g1;
  g1[0] = (int)(1u << 16);                               // data_size=1 (2 bytes)
  g1[1] = (int)((tensor_d0 & 0xffffu) << 16);            // tensor_dim0[15:0]
  g1[2] = (int)(((tensor_d0 >> 16) & 0xffffu)
        |       ((tensor_d1 & 0xffffu) << 16));          // td0[31:16] | td1[15:0]
  g1[3] = (int)(((tensor_d1 >> 16) & 0xffffu)
        |       ((tile_d0 & 0xffffu) << 16));            // td1[31:16] | tile_dim0
  g1[4] = (int)(tile_d1 & 0xffffu);                      // tile_dim1 (tile_dim2=0)
  g1[5] = (int)(unsigned int)(stride0_elems & 0xffffffffu);
  g1[6] = (int)(unsigned int)((stride0_elems >> 32) & 0xffffu);
  g1[7] = 0;
  i32x4 z4 = {0, 0, 0, 0};
#if __clang_major__ >= 23
  i32x8 z8 = {0, 0, 0, 0, 0, 0, 0, 0};
  __builtin_amdgcn_tensor_load_to_lds(g0, g1, z4, z4, z8, 0);
#else
  __builtin_amdgcn_tensor_load_to_lds(g0, g1, z4, z4, 0);
#endif
}

// ---------------------------------------------------------------------------
// Generic bf16 GEMM:  Out[M,N] = A[M,K] * Bt[N,K]^T + bias[N]
// Block: 256 threads (8 waves). Block tile 128x128, K-step 32.
// Wave tile 64x32 -> 4 Mx16 x 2 Nx16 accumulators, 8 WMMA per K-step.
// LDS tiles double-buffered, filled by TDM (wave 0 issues, TENSORcnt waits).
// OUT_MODE: 0 = f32 row-major, 1 = bf16 row-major, 2 = bf16 scattered into
//           Vt[b][h][hd][t] (attention V transpose, M=B*TK, N=D).
// ---------------------------------------------------------------------------
template<int OUT_MODE>
__global__ __launch_bounds__(256)
void gemm_bf16_wmma(const bf16* __restrict__ A, const bf16* __restrict__ Bt,
                    const float* __restrict__ bias, void* __restrict__ Out,
                    int M, int N, int K)
{
  __shared__ __align__(16) bf16 As[2][128 * 32];
  __shared__ __align__(16) bf16 Bs[2][128 * 32];

  const int tid  = threadIdx.x;
  const int lane = tid & 31;
  const int wave = tid >> 5;
  const int wm   = wave >> 2;      // 0..1  (M)
  const int wn   = wave & 3;       // 0..3  (N)
  const int bm   = blockIdx.y * 128;
  const int bn   = blockIdx.x * 128;
  const int l15  = lane & 15;
  const int aoff = (lane & 16) ? 8 : 0;    // A-frag K-chunk base (ISA 16-bit A layout)
  const int boff = (lane & 16) ? 8 : 0;    // B-frag K-chunk base (ISA 16-bit B layout)

  const v8f z8 = {0.f,0.f,0.f,0.f,0.f,0.f,0.f,0.f};
  v8f acc[4][2];
#pragma unroll
  for (int mt = 0; mt < 4; ++mt)
#pragma unroll
    for (int nt = 0; nt < 2; ++nt) acc[mt][nt] = z8;

  const unsigned int ldsA0 = (unsigned int)(size_t)&As[0][0];
  const unsigned int ldsA1 = (unsigned int)(size_t)&As[1][0];
  const unsigned int ldsB0 = (unsigned int)(size_t)&Bs[0][0];
  const unsigned int ldsB1 = (unsigned int)(size_t)&Bs[1][0];

  // Prime buffer 0 (one wave drives the Tensor Data Mover).
  if (wave == 0) {
    tdm_load_2d(ldsA0, A  + (size_t)bm * K, K, (unsigned)(M - bm), 32, 128, (unsigned long long)K);
    tdm_load_2d(ldsB0, Bt + (size_t)bn * K, K, (unsigned)(N - bn), 32, 128, (unsigned long long)K);
  }

  int buf = 0;
  for (int kb = 0; kb < K; kb += 32) {
    if (wave == 0) {
      if (kb + 32 < K) {  // prefetch next K tile into the other buffer
        tdm_load_2d(buf ? ldsA0 : ldsA1, A  + (size_t)bm * K + (kb + 32),
                    K, (unsigned)(M - bm), 32, 128, (unsigned long long)K);
        tdm_load_2d(buf ? ldsB0 : ldsB1, Bt + (size_t)bn * K + (kb + 32),
                    K, (unsigned)(N - bn), 32, 128, (unsigned long long)K);
        __builtin_amdgcn_s_wait_tensorcnt(2);   // current pair complete
      } else {
        __builtin_amdgcn_s_wait_tensorcnt(0);
      }
    }
    __syncthreads();

    v16bf bf[2];
#pragma unroll
    for (int nt = 0; nt < 2; ++nt) {
      const bf16* p = &Bs[buf][(wn * 32 + nt * 16 + l15) * 32 + boff];
      bf[nt] = make_frag(p, p + 16);
    }
#pragma unroll
    for (int mt = 0; mt < 4; ++mt) {
      const bf16* p = &As[buf][(wm * 64 + mt * 16 + l15) * 32 + aoff];
      v16bf af = make_frag(p, p + 16);
#pragma unroll
      for (int nt = 0; nt < 2; ++nt)
        acc[mt][nt] = __builtin_amdgcn_wmma_f32_16x16x32_bf16(
            false, af, false, bf[nt], (short)0, acc[mt][nt], false, false);
    }
    __syncthreads();
    buf ^= 1;
  }

  const int hi8 = (lane >> 4) << 3;
#pragma unroll
  for (int mt = 0; mt < 4; ++mt)
#pragma unroll
    for (int nt = 0; nt < 2; ++nt)
#pragma unroll
      for (int r = 0; r < 8; ++r) {
        int m = bm + wm * 64 + mt * 16 + hi8 + r;
        int n = bn + wn * 32 + nt * 16 + l15;
        float v = acc[mt][nt][r];
        if (bias) v += bias[n];
        if (OUT_MODE == 0) {
          ((float*)Out)[(size_t)m * N + n] = v;
        } else if (OUT_MODE == 1) {
          ((bf16*)Out)[(size_t)m * N + n] = (bf16)v;
        } else {
          int bb = m >> 11, t = m & (TK_ - 1);
          int hh = n >> 6,  hd = n & (HD_ - 1);
          ((bf16*)Out)[((size_t)((bb * H_ + hh) * HD_ + hd)) * TK_ + t] = (bf16)v;
        }
      }
}

// ---------------------------------------------------------------------------
// Flash-style attention. Block = 128 thr (4 waves), each wave owns 16 q rows.
// Grid: (TQ/64, H, B). Two passes (active keys / inactive keys), online
// softmax, P re-fragmented through a per-wave LDS tile, PV against Vt.
// ---------------------------------------------------------------------------
__global__ __launch_bounds__(128)
void attn_wmma(const bf16* __restrict__ Qp, const bf16* __restrict__ Km,
               const bf16* __restrict__ Vt, const float* __restrict__ log_temp,
               bf16* __restrict__ Ao)
{
  __shared__ __align__(16) bf16 Ps[4][16 * 32];

  const int tid  = threadIdx.x;
  const int lane = tid & 31;
  const int wave = tid >> 5;
  const int b    = blockIdx.z;
  const int h    = blockIdx.y;
  const int q0   = blockIdx.x * 64 + wave * 16;
  const int l15  = lane & 15;
  const int hi8  = (lane >> 4) << 3;
  const int aoff = (lane & 16) ? 8 : 0;
  const int boff = (lane & 16) ? 8 : 0;

  float temp = __expf(*log_temp);
  temp = fminf(fmaxf(temp, 0.1f), 10.0f);
  const float scale = 1.0f / ((float)HD_ * temp);

  // Q fragments resident for the whole key loop (K = 64 -> 2 x K32 chunks)
  v16bf qf[2];
  {
    const bf16* qrow = Qp + (size_t)(b * TQ_ + q0 + l15) * D_ + h * HD_;
#pragma unroll
    for (int kc = 0; kc < 2; ++kc) {
      const bf16* p = qrow + kc * 32 + aoff;
      qf[kc] = make_frag(p, p + 16);
    }
  }

  const v8f z8 = {0.f,0.f,0.f,0.f,0.f,0.f,0.f,0.f};
  v8f   acc[4];
  float m_run[8], l_run[8], resA[4][8];
  bf16* ps = &Ps[wave][0];

  for (int phase = 0; phase < 2; ++phase) {
    const int k_begin = phase ? NA_ : 0;
    const int k_end   = phase ? TK_ : NA_;
#pragma unroll
    for (int t = 0; t < 4; ++t) acc[t] = z8;
#pragma unroll
    for (int r = 0; r < 8; ++r) { m_run[r] = -1e30f; l_run[r] = 0.0f; }

    for (int kb = k_begin; kb < k_end; kb += 32) {
      // ---- S = (Q K^T) * scale over 32 keys (2 N-tiles x 2 K-chunks) ----
      v8f s[2]; s[0] = z8; s[1] = z8;
#pragma unroll
      for (int nt = 0; nt < 2; ++nt) {
        const bf16* krow = Km + (size_t)(b * TK_ + kb + nt * 16 + l15) * D_ + h * HD_;
#pragma unroll
        for (int kc = 0; kc < 2; ++kc) {
          const bf16* p = krow + kc * 32 + boff;
          v16bf kf = make_frag(p, p + 16);
          s[nt] = __builtin_amdgcn_wmma_f32_16x16x32_bf16(
              false, qf[kc], false, kf, (short)0, s[nt], false, false);
        }
      }
#pragma unroll
      for (int nt = 0; nt < 2; ++nt)
#pragma unroll
        for (int r = 0; r < 8; ++r) s[nt][r] *= scale;

      // ---- online softmax (row stats via 16-lane shfl reductions) ----
      float mnew[8], ef[8];
#pragma unroll
      for (int r = 0; r < 8; ++r) {
        float v = fmaxf(s[0][r], s[1][r]);
#pragma unroll
        for (int off = 1; off < 16; off <<= 1) v = fmaxf(v, __shfl_xor(v, off, 32));
        mnew[r] = fmaxf(m_run[r], v);
        ef[r]   = __expf(m_run[r] - mnew[r]);
        m_run[r] = mnew[r];
      }
#pragma unroll
      for (int r = 0; r < 8; ++r) {
        float p0 = __expf(s[0][r] - mnew[r]);
        float p1 = __expf(s[1][r] - mnew[r]);
        s[0][r] = p0; s[1][r] = p1;
        float rs = p0 + p1;
#pragma unroll
        for (int off = 1; off < 16; off <<= 1) rs += __shfl_xor(rs, off, 32);
        l_run[r] = l_run[r] * ef[r] + rs;
      }
#pragma unroll
      for (int t = 0; t < 4; ++t)
#pragma unroll
        for (int r = 0; r < 8; ++r) acc[t][r] *= ef[r];

      // ---- P (C-layout) -> LDS -> A-fragment ----
#pragma unroll
      for (int r = 0; r < 8; ++r) {
        ps[(hi8 + r) * 32 + l15]      = (bf16)s[0][r];
        ps[(hi8 + r) * 32 + 16 + l15] = (bf16)s[1][r];
      }
      __syncthreads();
      v16bf pf;
      {
        const bf16* p = &ps[l15 * 32 + aoff];
        pf = make_frag(p, p + 16);
      }
      // ---- acc += P * V  (4 hd-tiles, K = 32 keys) ----
#pragma unroll
      for (int ot = 0; ot < 4; ++ot) {
        const bf16* vrow = Vt + ((size_t)((b * H_ + h) * HD_ + ot * 16 + l15)) * TK_
                              + kb + boff;
        v16bf vf = make_frag(vrow, vrow + 16);
        acc[ot] = __builtin_amdgcn_wmma_f32_16x16x32_bf16(
            false, pf, false, vf, (short)0, acc[ot], false, false);
      }
      __syncthreads();
    }

    if (phase == 0) {
      const float inv_na = rsqrtf((float)NA_);
#pragma unroll
      for (int t = 0; t < 4; ++t)
#pragma unroll
        for (int r = 0; r < 8; ++r) resA[t][r] = acc[t][r] / l_run[r] * inv_na;
    } else {
      const float inv_ni = rsqrtf((float)NI_);
#pragma unroll
      for (int t = 0; t < 4; ++t)
#pragma unroll
        for (int r = 0; r < 8; ++r) {
          float o = resA[t][r] - acc[t][r] / l_run[r] * inv_ni;
          Ao[(size_t)(b * TQ_ + q0 + hi8 + r) * D_ + h * HD_ + t * 16 + l15] = (bf16)o;
        }
    }
  }
}

// ---------------------------------------------------------------------------
// Elementwise / norm / small kernels
// ---------------------------------------------------------------------------
__global__ __launch_bounds__(256)
void norm_kv_kernel(const float* __restrict__ kv, const float* __restrict__ kva_w,
                    const float* __restrict__ kvi_w, const float* __restrict__ fn_w,
                    const float* __restrict__ ab, const float* __restrict__ ib,
                    bf16* __restrict__ kvb, bf16* __restrict__ x)
{
  __shared__ float red[256];
  const int row = blockIdx.x;
  const int t   = row & (TK_ - 1);
  const float* src = kv + (size_t)row * D_;
  float v[4], ss = 0.f;
#pragma unroll
  for (int i = 0; i < 4; ++i) { v[i] = src[threadIdx.x + i * 256]; ss += v[i] * v[i]; }
  red[threadIdx.x] = ss; __syncthreads();
  for (int s2 = 128; s2 > 0; s2 >>= 1) {
    if (threadIdx.x < s2) red[threadIdx.x] += red[threadIdx.x + s2];
    __syncthreads();
  }
  const float rn = rsqrtf(red[0] / (float)D_ + 1e-6f);
  const float* w  = (t < NA_) ? kva_w : kvi_w;
  const float* bb = (t < NA_) ? ab    : ib;
#pragma unroll
  for (int i = 0; i < 4; ++i) {
    int d = threadIdx.x + i * 256;
    float nv = v[i] * rn;
    kvb[(size_t)row * D_ + d] = (bf16)(nv * w[d] + bb[d]);
    x  [(size_t)row * D_ + d] = (bf16)(nv * fn_w[d]);
  }
}

__global__ __launch_bounds__(256)
void norm_q_kernel(const float* __restrict__ q, const float* __restrict__ qn_w,
                   bf16* __restrict__ qn)
{
  __shared__ float red[256];
  const int row = blockIdx.x;
  const float* src = q + (size_t)row * D_;
  float v[4], ss = 0.f;
#pragma unroll
  for (int i = 0; i < 4; ++i) { v[i] = src[threadIdx.x + i * 256]; ss += v[i] * v[i]; }
  red[threadIdx.x] = ss; __syncthreads();
  for (int s2 = 128; s2 > 0; s2 >>= 1) {
    if (threadIdx.x < s2) red[threadIdx.x] += red[threadIdx.x + s2];
    __syncthreads();
  }
  const float rn = rsqrtf(red[0] / (float)D_ + 1e-6f);
#pragma unroll
  for (int i = 0; i < 4; ++i) {
    int d = threadIdx.x + i * 256;
    qn[(size_t)row * D_ + d] = (bf16)(v[i] * rn * qn_w[d]);
  }
}

__global__ __launch_bounds__(256)
void qnew_kernel(const float* __restrict__ q, const float* __restrict__ dp,
                 const float* __restrict__ dn_w, const float* __restrict__ gate_attn,
                 float* __restrict__ qout)
{
  __shared__ float red[256];
  const int row = blockIdx.x;
  const float* src = dp + (size_t)row * D_;
  float v[4], ss = 0.f;
#pragma unroll
  for (int i = 0; i < 4; ++i) { v[i] = src[threadIdx.x + i * 256]; ss += v[i] * v[i]; }
  red[threadIdx.x] = ss; __syncthreads();
  for (int s2 = 128; s2 > 0; s2 >>= 1) {
    if (threadIdx.x < s2) red[threadIdx.x] += red[threadIdx.x + s2];
    __syncthreads();
  }
  const float rn  = rsqrtf(red[0] / (float)D_ + 1e-6f);
  const float sga = sigmoidf_(*gate_attn);
#pragma unroll
  for (int i = 0; i < 4; ++i) {
    int d = threadIdx.x + i * 256;
    qout[(size_t)row * D_ + d] = q[(size_t)row * D_ + d] + sga * (v[i] * rn * dn_w[d]);
  }
}

__global__ __launch_bounds__(256)
void qvec_kernel(const float* __restrict__ qout, float* __restrict__ qv)
{
  const int idx = blockIdx.x * 256 + threadIdx.x;   // B_*D_ total
  const int b = idx >> 10, d = idx & (D_ - 1);
  float s = 0.f;
  for (int t = 0; t < TQ_; ++t) s += qout[(size_t)(b * TQ_ + t) * D_ + d];
  qv[idx] = s * (1.0f / (float)TQ_);
}

__global__ __launch_bounds__(256)
void router_kernel(const bf16* __restrict__ x, const float* __restrict__ qv,
                   const float* __restrict__ Wr, const float* __restrict__ br,
                   float* __restrict__ probs)
{
  __shared__ float red[256];
  __shared__ float outv[5];
  const int row = blockIdx.x;
  const int b   = row / TK_;
  float p[5] = {0.f, 0.f, 0.f, 0.f, 0.f};     // [0]=sim, [1..4]=logits
#pragma unroll
  for (int i = 0; i < 4; ++i) {
    int d = threadIdx.x + i * 256;
    float xv = (float)x[(size_t)row * D_ + d];
    p[0] += xv * qv[b * D_ + d];
#pragma unroll
    for (int e = 0; e < 4; ++e) p[1 + e] += xv * Wr[d * NE_ + e];
  }
  for (int qi = 0; qi < 5; ++qi) {
    red[threadIdx.x] = p[qi]; __syncthreads();
    for (int s2 = 128; s2 > 0; s2 >>= 1) {
      if (threadIdx.x < s2) red[threadIdx.x] += red[threadIdx.x + s2];
      __syncthreads();
    }
    if (threadIdx.x == 0) outv[qi] = red[0];
    __syncthreads();
  }
  if (threadIdx.x == 0) {
    float sim = outv[0];
    float lg[4], mx = -1e30f;
    for (int e = 0; e < 4; ++e) { lg[e] = outv[1 + e] + br[e] + sim; mx = fmaxf(mx, lg[e]); }
    float ex[4], sum = 0.f;
    for (int e = 0; e < 4; ++e) { ex[e] = __expf(lg[e] - mx); sum += ex[e]; }
    for (int e = 0; e < 4; ++e) probs[row * NE_ + e] = ex[e] / sum;
  }
}

__global__ __launch_bounds__(128)
void act_kernel(const float* __restrict__ h, const float* __restrict__ probs,
                bf16* __restrict__ act)
{
  const int row = blockIdx.x;
  const int j = threadIdx.x;            // 0..127
  const int e = j >> 5, i = j & 31;
  float xp = h[(size_t)row * (NE_ * 2 * HID_) + e * 64 + i];
  float g  = h[(size_t)row * (NE_ * 2 * HID_) + e * 64 + 32 + i];
  float a  = xp * g * sigmoidf_(g);
  act[(size_t)row * (NE_ * HID_) + j] = (bf16)(a * probs[row * NE_ + e]);
}

__global__ __launch_bounds__(256)
void kvnew_kernel(const float* __restrict__ kv, const float* __restrict__ moe,
                  const float* __restrict__ probs, const float* __restrict__ b2,
                  const float* __restrict__ gate_ffn, float* __restrict__ kvout)
{
  const int row = blockIdx.x;
  const float sgf = sigmoidf_(*gate_ffn);
  const float p0 = probs[row * 4 + 0], p1 = probs[row * 4 + 1];
  const float p2 = probs[row * 4 + 2], p3 = probs[row * 4 + 3];
#pragma unroll
  for (int i = 0; i < 4; ++i) {
    int d = threadIdx.x + i * 256;
    float bb = p0 * b2[d] + p1 * b2[D_ + d] + p2 * b2[2 * D_ + d] + p3 * b2[3 * D_ + d];
    kvout[(size_t)row * D_ + d] =
        kv[(size_t)row * D_ + d] + sgf * (moe[(size_t)row * D_ + d] + bb);
  }
}

// ---- weight prep (f32 -> bf16 Bt layouts) ----
__global__ void transpose_sq_kernel(const float* __restrict__ W, bf16* __restrict__ Wt) {
  int idx = blockIdx.x * 256 + threadIdx.x;      // n*1024 + k
  int n = idx >> 10, k = idx & 1023;
  Wt[idx] = (bf16)W[k * 1024 + n];
}
__global__ void prep_w1_kernel(const float* __restrict__ W1, bf16* __restrict__ W1t) {
  int id = blockIdx.x * 256 + threadIdx.x;       // 256*1024 total
  int n = id >> 10, k = id & 1023;
  int e = n >> 6, j = n & 63;
  W1t[id] = (bf16)W1[((size_t)e * D_ + k) * (2 * HID_) + j];
}
__global__ void prep_w2_kernel(const float* __restrict__ W2, bf16* __restrict__ W2t) {
  int id = blockIdx.x * 256 + threadIdx.x;       // 1024*128 total
  int n = id >> 7, c = id & 127;
  int e = c >> 5, kk = c & 31;
  W2t[id] = (bf16)W2[((size_t)e * HID_ + kk) * D_ + n];
}

// ---------------------------------------------------------------------------
// Host-side orchestration
// ---------------------------------------------------------------------------
extern "C" void kernel_launch(void* const* d_in, const int* in_sizes, int n_in,
                              void* d_out, int out_size, void* d_ws, size_t ws_size,
                              hipStream_t stream)
{
  (void)in_sizes; (void)n_in; (void)out_size; (void)ws_size;

  const float* q        = (const float*)d_in[0];
  const float* kv       = (const float*)d_in[1];
  const float* qn_w     = (const float*)d_in[2];
  const float* kva_w    = (const float*)d_in[3];
  const float* kvi_w    = (const float*)d_in[4];
  const float* dn_w     = (const float*)d_in[5];
  const float* fn_w     = (const float*)d_in[6];
  const float* Wq       = (const float*)d_in[7];
  const float* bq       = (const float*)d_in[8];
  const float* Wk       = (const float*)d_in[9];
  const float* bk       = (const float*)d_in[10];
  const float* Wv       = (const float*)d_in[11];
  const float* bv       = (const float*)d_in[12];
  const float* Wo       = (const float*)d_in[13];
  const float* bo       = (const float*)d_in[14];
  const float* ab       = (const float*)d_in[15];
  const float* ib       = (const float*)d_in[16];
  const float* log_temp = (const float*)d_in[17];
  const float* gate_attn= (const float*)d_in[18];
  const float* gate_ffn = (const float*)d_in[19];
  const float* Wr       = (const float*)d_in[20];
  const float* br       = (const float*)d_in[21];
  const float* W1       = (const float*)d_in[22];
  const float* b1       = (const float*)d_in[23];
  const float* W2       = (const float*)d_in[24];
  const float* b2       = (const float*)d_in[25];

  const size_t MBy = 1024ull * 1024ull;
  char* ws = (char*)d_ws;
  bf16*  Wqt  = (bf16*)(ws + 0 * MBy);          // 2 MB each
  bf16*  Wkt  = (bf16*)(ws + 2 * MBy);
  bf16*  Wvt  = (bf16*)(ws + 4 * MBy);
  bf16*  Wot  = (bf16*)(ws + 6 * MBy);
  bf16*  W1t  = (bf16*)(ws + 8 * MBy);          // 512 KB
  bf16*  W2t  = (bf16*)(ws + 8 * MBy + 512 * 1024);  // 256 KB
  bf16*  qn   = (bf16*)(ws + 9 * MBy);          // 8 MB
  bf16*  kvb  = (bf16*)(ws + 17 * MBy);         // 32 MB
  bf16*  xn   = (bf16*)(ws + 49 * MBy);         // 32 MB
  bf16*  qp   = (bf16*)(ws + 81 * MBy);         // 8 MB
  bf16*  km   = (bf16*)(ws + 89 * MBy);         // 32 MB
  bf16*  vt   = (bf16*)(ws + 121 * MBy);        // 32 MB
  bf16*  ao   = (bf16*)(ws + 153 * MBy);        // 8 MB
  float* dp   = (float*)(ws + 161 * MBy);       // 16 MB
  float* qv   = (float*)(ws + 177 * MBy);       // 32 KB
  float* probs= (float*)(ws + 177 * MBy + 64 * 1024); // 256 KB
  float* hbuf = (float*)(ws + 178 * MBy);       // 16 MB
  bf16*  act  = (bf16*)(ws + 194 * MBy);        // 4 MB
  float* moe  = (float*)(ws + 198 * MBy);       // 64 MB

  float* qout  = (float*)d_out;
  float* kvout = (float*)d_out + (size_t)B_ * TQ_ * D_;

  const int MQ = B_ * TQ_;   // 4096
  const int MK = B_ * TK_;   // 16384

  // 1) weight prep
  transpose_sq_kernel<<<4096, 256, 0, stream>>>(Wq, Wqt);
  transpose_sq_kernel<<<4096, 256, 0, stream>>>(Wk, Wkt);
  transpose_sq_kernel<<<4096, 256, 0, stream>>>(Wv, Wvt);
  transpose_sq_kernel<<<4096, 256, 0, stream>>>(Wo, Wot);
  prep_w1_kernel<<<1024, 256, 0, stream>>>(W1, W1t);
  prep_w2_kernel<<<512, 256, 0, stream>>>(W2, W2t);

  // 2) norms -> bf16 activations
  norm_q_kernel<<<MQ, 256, 0, stream>>>(q, qn_w, qn);
  norm_kv_kernel<<<MK, 256, 0, stream>>>(kv, kva_w, kvi_w, fn_w, ab, ib, kvb, xn);

  // 3) projections (bf16 WMMA GEMMs, TDM-fed)
  gemm_bf16_wmma<1><<<dim3(8, 32), 256, 0, stream>>>(qn, Wqt, bq, qp, MQ, D_, D_);
  gemm_bf16_wmma<1><<<dim3(8, 128), 256, 0, stream>>>(kvb, Wkt, bk, km, MK, D_, D_);
  gemm_bf16_wmma<2><<<dim3(8, 128), 256, 0, stream>>>(kvb, Wvt, bv, vt, MK, D_, D_);

  // 4) attention
  attn_wmma<<<dim3(TQ_ / 64, H_, B_), 128, 0, stream>>>(qp, km, vt, log_temp, ao);

  // 5) O projection + residual/gate + q_vec
  gemm_bf16_wmma<0><<<dim3(8, 32), 256, 0, stream>>>(ao, Wot, bo, dp, MQ, D_, D_);
  qnew_kernel<<<MQ, 256, 0, stream>>>(q, dp, dn_w, gate_attn, qout);
  qvec_kernel<<<(B_ * D_) / 256, 256, 0, stream>>>(qout, qv);

  // 6) router + MoE (experts fused: probs folded into activations -> one K=128 GEMM)
  router_kernel<<<MK, 256, 0, stream>>>(xn, qv, Wr, br, probs);
  gemm_bf16_wmma<0><<<dim3(2, 128), 256, 0, stream>>>(xn, W1t, b1, hbuf, MK, NE_ * 2 * HID_, D_);
  act_kernel<<<MK, 128, 0, stream>>>(hbuf, probs, act);
  gemm_bf16_wmma<0><<<dim3(8, 128), 256, 0, stream>>>(act, W2t, nullptr, moe, MK, D_, NE_ * HID_);
  kvnew_kernel<<<MK, 256, 0, stream>>>(kv, moe, probs, b2, gate_ffn, kvout);
}
